// RNN_75548474737189
// MI455X (gfx1250) — matrix operand order repaired and checked by reading, not compiled, not run on hardware
//
#include <hip/hip_runtime.h>
#include <math.h>

// ---------------------------------------------------------------------------
// Fused 2-layer LSTM (H=64, IN=28, B=2048, T=512) + final projection, CDNA5.
//   - weights staged once into LDS as f16 (B-matrix [n][k] layout)
//   - per-timestep gate GEMMs via v_wmma_f32_16x16x32_f16 (28 WMMAs/step)
//   - each wave owns i/f/g/o tiles for a 16x16 (batch x unit) patch; LSTM
//     pointwise update runs in accumulator registers, cell state c persistent
//     in VGPRs for all 512 steps
//   - x(t+1) prefetched to registers (non-temporal) under GEMM0, committed to
//     LDS in the pointwise phase: 3 barriers/step
//   - transcendentals via v_exp_f32/v_rcp_f32 (no branchy ocml tanh)
// ---------------------------------------------------------------------------

typedef __attribute__((ext_vector_type(16))) _Float16 v16h;
typedef __attribute__((ext_vector_type(8)))  float    v8f;

#define B_TOTAL  2048
#define T_STEPS  512
#define IN_DIM   28
#define H_DIM    64
#define G_DIM    256      // 4*H
#define K0_PAD   96       // layer0 K: 28 (pad->32) + 64
#define K1_DIM   128      // layer1 K: 64 + 64
#define B_TILE   32
#define NTHREADS 256      // 8 waves (wave32)
#define XELEMS   (B_TILE * IN_DIM)   // 896

static __device__ inline float sigmoid_f(float v) {
  // 1/(1+e^-v): v_exp_f32 + v_rcp_f32, saturates correctly at +/-inf
  return __builtin_amdgcn_rcpf(1.0f + __expf(-v));
}

static __device__ inline float tanh_f(float v) {
  // tanh(v) = 1 - 2/(1+e^{2v}); e^{2v}->inf => 1, e^{2v}->0 => -1
  return 1.0f - 2.0f * __builtin_amdgcn_rcpf(1.0f + __expf(2.0f * v));
}

static __device__ inline v8f wmma_f16(v16h a, v16h b, v8f c) {
  // D = A(16x32 f16) * B(32x16 f16) + C(16x16 f32)
  return __builtin_amdgcn_wmma_f32_16x16x32_f16(false, a, false, b,
                                                (short)0, c, false, false);
}

// A fragment (16-bit A 16x32): lane half 0 holds K {0..7,16..23}, half 1 holds
// K {8..15,24..31} for row M = lane&15 (ISA 7.12.2).
static __device__ inline v16h load_a_frag(const _Float16* A, int ld, int row,
                                          int k0, int half) {
  v16h a;
  const _Float16* p = A + row * ld + k0 + half * 8;
#pragma unroll
  for (int e = 0; e < 8; ++e)  a[e] = p[e];
#pragma unroll
  for (int e = 0; e < 8; ++e)  a[8 + e] = p[16 + e];
  return a;
}

// B fragment (16-bit B 32x16): lane fixes N; lane half 0 holds K {0..7,16..23},
// half 1 holds K {8..15,24..31} (same order as A) in W[n][k].
static __device__ inline v16h load_b_frag(const _Float16* W, int ld, int n,
                                          int k0, int half) {
  v16h b;
  const _Float16* p = W + n * ld + k0 + half * 8;
#pragma unroll
  for (int e = 0; e < 8; ++e)  b[e] = p[e];
#pragma unroll
  for (int e = 0; e < 8; ++e)  b[8 + e] = p[16 + e];
  return b;
}

static __device__ inline v8f splat8(float v) {
  v8f r;
#pragma unroll
  for (int e = 0; e < 8; ++e) r[e] = v;
  return r;
}

__global__ __launch_bounds__(NTHREADS, 1) void lstm2_wmma(
    const float* __restrict__ x,
    const float* __restrict__ Wih0, const float* __restrict__ Whh0,
    const float* __restrict__ bih0, const float* __restrict__ bhh0,
    const float* __restrict__ Wih1, const float* __restrict__ Whh1,
    const float* __restrict__ bih1, const float* __restrict__ bhh1,
    const float* __restrict__ Wout, const float* __restrict__ bout,
    float* __restrict__ out) {
  __shared__ _Float16 sW0[G_DIM * K0_PAD];  // 48 KB  B-matrix [n][k], K padded
  __shared__ _Float16 sW1[G_DIM * K1_DIM];  // 64 KB
  __shared__ float    sB0[G_DIM];
  __shared__ float    sB1[G_DIM];
  __shared__ _Float16 sA0[B_TILE * K0_PAD]; // [x_t | h_l0]   6 KB
  __shared__ _Float16 sA1[B_TILE * K1_DIM]; // [h_l0 | h_l1]  8 KB
  __shared__ float    sHL[B_TILE * H_DIM];  // final h_l1 (f32) 8 KB
  __shared__ float    sWo[10 * H_DIM];
  __shared__ float    sBo[10];

  const int tid  = threadIdx.x;
  const int lane = tid & 31;
  const int wid  = tid >> 5;
  const int m    = wid >> 2;            // M tile (batch)  0..1
  const int j    = wid & 3;             // unit tile       0..3
  const int half = lane >> 4;
  const int nlo  = lane & 15;
  const int row0 = m * 16;
  const int ui   = j * 16 + nlo;        // hidden unit owned by this lane
  const int arow = row0 + nlo;          // A-fragment row for this lane

  // ---- stage weights/biases into LDS (f32 -> f16 for WMMA operands) ----
  for (int i = tid; i < G_DIM * K0_PAD; i += NTHREADS) {
    int n = i / K0_PAD, k = i % K0_PAD;
    float w = 0.0f;
    if (k < IN_DIM)   w = Wih0[n * IN_DIM + k];
    else if (k >= 32) w = Whh0[n * H_DIM + (k - 32)];
    sW0[i] = (_Float16)w;
  }
  for (int i = tid; i < G_DIM * K1_DIM; i += NTHREADS) {
    int n = i / K1_DIM, k = i % K1_DIM;
    float w = (k < H_DIM) ? Wih1[n * H_DIM + k] : Whh1[n * H_DIM + (k - H_DIM)];
    sW1[i] = (_Float16)w;
  }
  for (int i = tid; i < G_DIM; i += NTHREADS) {
    sB0[i] = bih0[i] + bhh0[i];
    sB1[i] = bih1[i] + bhh1[i];
  }
  for (int i = tid; i < 10 * H_DIM; i += NTHREADS) sWo[i] = Wout[i];
  if (tid < 10) sBo[tid] = bout[tid];
  for (int i = tid; i < B_TILE * K0_PAD; i += NTHREADS) sA0[i] = (_Float16)0.0f;
  for (int i = tid; i < B_TILE * K1_DIM; i += NTHREADS) sA1[i] = (_Float16)0.0f;

  const size_t b0 = (size_t)blockIdx.x * B_TILE;

  // x prefetch helpers: 896 elems / 256 threads -> up to 4 per thread
  float xbuf[4];
  auto load_x = [&](int t) {
#pragma unroll
    for (int q = 0; q < 4; ++q) {
      int idx = tid + q * NTHREADS;
      if (idx < XELEMS) {
        int r = idx / IN_DIM, k = idx % IN_DIM;
        xbuf[q] = __builtin_nontemporal_load(
            &x[((b0 + r) * T_STEPS + (size_t)t) * IN_DIM + k]);
      }
    }
  };
  auto commit_x = [&]() {
#pragma unroll
    for (int q = 0; q < 4; ++q) {
      int idx = tid + q * NTHREADS;
      if (idx < XELEMS) {
        int r = idx / IN_DIM, k = idx % IN_DIM;
        sA0[r * K0_PAD + k] = (_Float16)xbuf[q];
      }
    }
  };

  load_x(0);
  commit_x();
  __syncthreads();

  // hoist loop-invariant per-lane bias values out of the recurrence
  const float b0i = sB0[ui],       b0f = sB0[64 + ui];
  const float b0g = sB0[128 + ui], b0o = sB0[192 + ui];
  const float b1i = sB1[ui],       b1f = sB1[64 + ui];
  const float b1g = sB1[128 + ui], b1o = sB1[192 + ui];

  v8f c0 = splat8(0.0f);                // persistent cell states (registers)
  v8f c1 = splat8(0.0f);

  for (int t = 0; t < T_STEPS; ++t) {
    // issue x(t+1) global loads early; GEMM0's WMMAs cover the HBM latency
    if (t + 1 < T_STEPS) load_x(t + 1);

    // ---- layer 0 gates: [32 x 96] @ [96 x 256], bias folded into C ----
    v8f gi = splat8(b0i);
    v8f gf = splat8(b0f);
    v8f gg = splat8(b0g);
    v8f go = splat8(b0o);
#pragma unroll
    for (int ks = 0; ks < 3; ++ks) {
      v16h a = load_a_frag(sA0, K0_PAD, arow, ks * 32, half);
      gi = wmma_f16(a, load_b_frag(sW0, K0_PAD, ui,        ks * 32, half), gi);
      gf = wmma_f16(a, load_b_frag(sW0, K0_PAD, 64 + ui,   ks * 32, half), gf);
      gg = wmma_f16(a, load_b_frag(sW0, K0_PAD, 128 + ui,  ks * 32, half), gg);
      go = wmma_f16(a, load_b_frag(sW0, K0_PAD, 192 + ui,  ks * 32, half), go);
    }
    __syncthreads();  // all waves done reading sA0

    // pointwise LSTM update in accumulator registers (C/D layout: N=nlo,
    // M = e + 8*half); write h_l0 (f16) and commit prefetched x(t+1)
#pragma unroll
    for (int e = 0; e < 8; ++e) {
      float iv = sigmoid_f(gi[e]);
      float fv = sigmoid_f(gf[e]);
      float gv = tanh_f(gg[e]);
      float ov = sigmoid_f(go[e]);
      float cv = fv * c0[e] + iv * gv;
      c0[e] = cv;
      float hv = ov * tanh_f(cv);
      int r = row0 + e + half * 8;
      _Float16 hh = (_Float16)hv;
      sA1[r * K1_DIM + ui]      = hh;   // layer-1 input (this step)
      sA0[r * K0_PAD + 32 + ui] = hh;   // layer-0 recurrent input (next step)
    }
    if (t + 1 < T_STEPS) commit_x();
    __syncthreads();  // sA1[:,0:64] + next-step sA0 ready

    // ---- layer 1 gates: [32 x 128] @ [128 x 256] ----
    gi = splat8(b1i);
    gf = splat8(b1f);
    gg = splat8(b1g);
    go = splat8(b1o);
#pragma unroll
    for (int ks = 0; ks < 4; ++ks) {
      v16h a = load_a_frag(sA1, K1_DIM, arow, ks * 32, half);
      gi = wmma_f16(a, load_b_frag(sW1, K1_DIM, ui,        ks * 32, half), gi);
      gf = wmma_f16(a, load_b_frag(sW1, K1_DIM, 64 + ui,   ks * 32, half), gf);
      gg = wmma_f16(a, load_b_frag(sW1, K1_DIM, 128 + ui,  ks * 32, half), gg);
      go = wmma_f16(a, load_b_frag(sW1, K1_DIM, 192 + ui,  ks * 32, half), go);
    }
    __syncthreads();  // all waves done reading sA1

#pragma unroll
    for (int e = 0; e < 8; ++e) {
      float iv = sigmoid_f(gi[e]);
      float fv = sigmoid_f(gf[e]);
      float gv = tanh_f(gg[e]);
      float ov = sigmoid_f(go[e]);
      float cv = fv * c1[e] + iv * gv;
      c1[e] = cv;
      float hv = ov * tanh_f(cv);
      int r = row0 + e + half * 8;
      sA1[r * K1_DIM + 64 + ui] = (_Float16)hv;  // recurrent input next step
      if (t == T_STEPS - 1) sHL[r * H_DIM + ui] = hv;
    }
    // next iteration's post-GEMM0 barrier orders these writes before reuse
  }
  __syncthreads();

  // ---- final projection: out = h_last @ Wout^T + bout (tiny, plain VALU) ----
  for (int i = tid; i < B_TILE * 10; i += NTHREADS) {
    int b = i / 10, o = i % 10;
    float s = sBo[o];
#pragma unroll
    for (int u = 0; u < H_DIM; ++u) s += sHL[b * H_DIM + u] * sWo[o * H_DIM + u];
    out[(b0 + b) * 10 + o] = s;
  }
  for (int i = tid; i < B_TILE * H_DIM; i += NTHREADS) {
    int b = i / H_DIM, u = i % H_DIM;
    out[(size_t)B_TOTAL * 10 + (b0 + b) * H_DIM + u] = sHL[i];
  }
}

extern "C" void kernel_launch(void* const* d_in, const int* in_sizes, int n_in,
                              void* d_out, int out_size, void* d_ws,
                              size_t ws_size, hipStream_t stream) {
  (void)in_sizes; (void)n_in; (void)out_size; (void)d_ws; (void)ws_size;
  dim3 grid(B_TOTAL / B_TILE);   // 64 workgroups
  dim3 block(NTHREADS);          // 8 wave32 waves
  lstm2_wmma<<<grid, block, 0, stream>>>(
      (const float*)d_in[0],  (const float*)d_in[1], (const float*)d_in[2],
      (const float*)d_in[3],  (const float*)d_in[4], (const float*)d_in[5],
      (const float*)d_in[6],  (const float*)d_in[7], (const float*)d_in[8],
      (const float*)d_in[9],  (const float*)d_in[10], (float*)d_out);
}
